// MS_Attention_linear_3d_15633680957882
// MI455X (gfx1250) — compile-verified, matrix operand order checked by reading, not run on hardware
//
#include <hip/hip_runtime.h>

// ---------------------------------------------------------------------------
// MS_Attention_linear_3d — spiking linear attention, CDNA5 (gfx1250) WMMA.
// All heavy GEMMs run on v_wmma_f32_16x16x32_f16: every GEMM here has at
// least one binary {0,1} spike operand (exact in f16).
// Spike & pre-activation tensors live in workspace in n-major [batch][N][C]
// layout so all hot-loop tile fills are aligned 16B async-to-LDS copies and
// no LDS transpose is needed in the big GEMMs. Main GEMM: BM=256 x BN=64,
// 8 waves x (4x2 WMMA tiles) = 8 v_wmma per K-step per wave.
// ---------------------------------------------------------------------------

typedef _Float16 v16h __attribute__((ext_vector_type(16)));
typedef _Float16 v8h  __attribute__((ext_vector_type(8)));
typedef float    v8f  __attribute__((ext_vector_type(8)));
typedef int      v4i  __attribute__((vector_size(16)));

#define TT    4
#define BB    8
#define CCH   512
#define NPIX  576
#define HEADS 8
#define DH    64

// ---- async global->LDS (CDNA5), with portable fallback ---------------------
#if defined(__has_builtin)
#  if __has_builtin(__builtin_amdgcn_global_load_async_to_lds_b128)
#    define HAVE_ASYNC_LDS 1
#  endif
#endif
#ifndef HAVE_ASYNC_LDS
#  define HAVE_ASYNC_LDS 0
#endif

__device__ __forceinline__ void cp16(_Float16* l, const _Float16* g) {
#if HAVE_ASYNC_LDS
    __builtin_amdgcn_global_load_async_to_lds_b128(
        (__attribute__((address_space(1))) v4i*)(const_cast<_Float16*>(g)),
        (__attribute__((address_space(3))) v4i*)(l), 0, 0);
#else
    *(v8h*)l = *(const v8h*)g;
#endif
}

__device__ __forceinline__ void cp_wait() {
#if HAVE_ASYNC_LDS
#  if __has_builtin(__builtin_amdgcn_s_wait_asynccnt)
    __builtin_amdgcn_s_wait_asynccnt(0);
#  else
    asm volatile("s_wait_asynccnt 0" ::: "memory");
#  endif
#endif
}

// ---- WMMA fragment helpers -------------------------------------------------
// 16-bit A-matrix layout (ISA 7.12.2): lane 0-15 row M=lane holds K={0..7,16..23},
// lane 16-31 row M=lane-16 holds K={8..15,24..31}. B stored n-major in LDS so the
// identical loader serves both operands. Row pitch LD is a 16B multiple.
__device__ __forceinline__ v16h load_frag(const _Float16* base, int row, int ld,
                                          int kbase, int lane) {
    const _Float16* p = base + row * ld + kbase + ((lane & 16) >> 1);
    v8h lo = *(const v8h*)(p);
    v8h hi = *(const v8h*)(p + 16);
    v16h r;
#pragma unroll
    for (int i = 0; i < 8; ++i) { r[i] = lo[i]; r[i + 8] = hi[i]; }
    return r;
}

__device__ __forceinline__ v8f wmma_f16(v16h a, v16h b, v8f c) {
    return __builtin_amdgcn_wmma_f32_16x16x32_f16(false, a, false, b,
                                                  (short)0, c, false, false);
}

// ---- small utility kernels -------------------------------------------------
__global__ void cvt_f32_f16(const float* __restrict__ in, _Float16* __restrict__ out, int n) {
    int i = blockIdx.x * blockDim.x + threadIdx.x;
    if (i < n) out[i] = (_Float16)in[i];
}

// LIF over input x[t,b,c,n] + pos[c,n] -> spikes (f16 {0,1}) in [b][n][c] layout.
__global__ void lif_input(const float* __restrict__ x, const float* __restrict__ pos,
                          _Float16* __restrict__ spk) {
    const int stride = BB * NPIX * CCH;
    int i = blockIdx.x * blockDim.x + threadIdx.x;
    if (i >= stride) return;
    int c = i % CCH;
    int n = (i / CCH) % NPIX;
    int b = i / (CCH * NPIX);
    const float p = pos[c * NPIX + n];
    float mem = 0.f, s = 0.f;
#pragma unroll
    for (int t = 0; t < TT; ++t) {
        mem = mem * 0.25f * (1.f - s) + x[(((size_t)t * BB + b) * CCH + c) * NPIX + n] + p;
        s = (mem > 0.5f) ? 1.f : 0.f;
        spk[(size_t)t * stride + i] = (_Float16)s;
    }
}

// LIF over f32 pre-activations, both in [t][b][n][c] layout (fully linear).
__global__ void lif_scan(const float* __restrict__ pre, _Float16* __restrict__ spk) {
    const int stride = BB * NPIX * CCH;
    int i = blockIdx.x * blockDim.x + threadIdx.x;
    if (i >= stride) return;
    float mem = 0.f, s = 0.f;
#pragma unroll
    for (int t = 0; t < TT; ++t) {
        mem = mem * 0.25f * (1.f - s) + pre[(size_t)t * stride + i];
        s = (mem > 0.5f) ? 1.f : 0.f;
        spk[(size_t)t * stride + i] = (_Float16)s;
    }
}

// ---- main projection GEMM --------------------------------------------------
// out[m][n] = (sum_k W[m][k] * Sp[n][k]) * sc[m] + bi[m]
// W: [512][512] row-major f16; Sp: [batch][576][512] spikes (n-major).
// Block tile 256x64, BK=32; wave tile 64x32 (4x2 WMMA tiles, 8 wmma/K-step).
// TOUT=1: write pre [batch][n][m] (c-contig, vector stores, feeds lif_scan)
// TOUT=0: write out [batch][m][n] (final d_out layout [T,B,C,H*W])
#define LDK 48   // 32 K halves + 16 pad (96B row pitch)
template <bool TOUT>
__global__ __launch_bounds__(256)
void gemm_spk(const _Float16* __restrict__ W16,
              const _Float16* __restrict__ Sp,
              const float* __restrict__ sc, const float* __restrict__ bi,
              float* __restrict__ out) {
    __shared__ _Float16 As[256 * LDK];   // 24 KB, [m][k]
    __shared__ _Float16 Bs[64  * LDK];   //  6 KB, [n][k]
    const int tid = threadIdx.x, lane = tid & 31, w = tid >> 5;
    const int m0 = blockIdx.y * 256, n0 = blockIdx.x * 64;
    const _Float16* Sb = Sp  + (size_t)blockIdx.z * NPIX * CCH;
    float*          Ob = out + (size_t)blockIdx.z * NPIX * CCH;

    const int wm = (w & 3) * 64, wn = (w >> 2) * 32;
    const int col0 = lane & 15, rbase = (lane & 16) ? 8 : 0;
    v8f acc[4][2] = {};

    // per-thread tile-fill coordinates (constant across K loop)
    const int bn = tid >> 2, bkc = (tid & 3) * 8;    // B: 64 rows x 16B chunks

    for (int k0 = 0; k0 < CCH; k0 += 32) {
        // A tile 256x32: each thread copies one full 64B row slice (4 async b128)
        const _Float16* ga = W16 + (size_t)(m0 + tid) * CCH + k0;
        _Float16*       la = &As[tid * LDK];
        cp16(la,      ga);
        cp16(la + 8,  ga + 8);
        cp16(la + 16, ga + 16);
        cp16(la + 24, ga + 24);
        // B tile 64x32: one async b128 per thread
        cp16(&Bs[bn * LDK + bkc], Sb + (size_t)(n0 + bn) * CCH + k0 + bkc);
        cp_wait();
        __syncthreads();
        v16h b0 = load_frag(Bs, wn + col0,      LDK, 0, lane);
        v16h b1 = load_frag(Bs, wn + 16 + col0, LDK, 0, lane);
#pragma unroll
        for (int mi = 0; mi < 4; ++mi) {
            v16h a = load_frag(As, wm + mi * 16 + col0, LDK, 0, lane);
            acc[mi][0] = wmma_f16(a, b0, acc[mi][0]);
            acc[mi][1] = wmma_f16(a, b1, acc[mi][1]);
        }
        __syncthreads();
    }
#pragma unroll
    for (int mi = 0; mi < 4; ++mi)
#pragma unroll
        for (int ni = 0; ni < 2; ++ni) {
            const int gn = n0 + wn + ni * 16 + col0;
            const int mstart = m0 + wm + mi * 16 + rbase;
            if (TOUT) {
                v8f o;
#pragma unroll
                for (int r = 0; r < 8; ++r)
                    o[r] = acc[mi][ni][r] * sc[mstart + r] + bi[mstart + r];
                *(v8f*)&Ob[(size_t)gn * CCH + mstart] = o;   // 32B-aligned
            } else {
#pragma unroll
                for (int r = 0; r < 8; ++r)
                    Ob[(size_t)(mstart + r) * NPIX + gn] =
                        acc[mi][ni][r] * sc[mstart + r] + bi[mstart + r];
            }
        }
}

// ---- kv = K^T V per (t,b,head): kv[d][e] = sum_n Ks[n][d]*Vs[n][e] ---------
// Spikes are [n][c]; the k-dim (n) is the row index, so fill transposes 64x32
// tiles into k-contiguous LDS (small kernel: 256 blocks, 18 steps).
__global__ __launch_bounds__(256)
void kv_gemm(const _Float16* __restrict__ Ksp, const _Float16* __restrict__ Vsp,
             float* __restrict__ kvout) {
    __shared__ _Float16 As[64 * LDK];   // [d][n-chunk]
    __shared__ _Float16 Bs[64 * LDK];   // [e][n-chunk]
    const int tid = threadIdx.x, lane = tid & 31, w = tid >> 5;
    const int tbh = blockIdx.x, tb = tbh >> 3, h = tbh & 7;
    const _Float16* Kb = Ksp + (size_t)tb * NPIX * CCH + h * DH;
    const _Float16* Vb = Vsp + (size_t)tb * NPIX * CCH + h * DH;

    const int wm = (w & 3) * 16, we = (w >> 2) * 32;
    const int col0 = lane & 15, rbase = (lane & 16) ? 8 : 0;
    v8f acc[2] = {};

    for (int n0 = 0; n0 < NPIX; n0 += 32) {
        int nn = tid >> 3, dch = (tid & 7) * 8;
        v8h gk = *(const v8h*)(Kb + (size_t)(n0 + nn) * CCH + dch);
        v8h gv = *(const v8h*)(Vb + (size_t)(n0 + nn) * CCH + dch);
#pragma unroll
        for (int j = 0; j < 8; ++j) {
            As[(dch + j) * LDK + nn] = gk[j];
            Bs[(dch + j) * LDK + nn] = gv[j];
        }
        __syncthreads();
        v16h a  = load_frag(As, wm + col0,      LDK, 0, lane);
        v16h b0 = load_frag(Bs, we + col0,      LDK, 0, lane);
        v16h b1 = load_frag(Bs, we + 16 + col0, LDK, 0, lane);
        acc[0] = wmma_f16(a, b0, acc[0]);
        acc[1] = wmma_f16(a, b1, acc[1]);
        __syncthreads();
    }
#pragma unroll
    for (int ei = 0; ei < 2; ++ei)
#pragma unroll
        for (int r = 0; r < 8; ++r)
            kvout[(size_t)tbh * DH * DH + (wm + rbase + r) * DH + we + ei * 16 + col0]
                = acc[ei][r];
}

// ---- attn pre: out[e][n] = 0.25 * sum_d kv[d][e] * Qs[n][d]; K = 64 --------
#define LDK2 80   // 64 K halves + 16 pad (160B row pitch)
__global__ __launch_bounds__(256)
void attn_gemm(const float* __restrict__ kvin, const _Float16* __restrict__ Qsp,
               float* __restrict__ pre) {
    __shared__ _Float16 As[64 * LDK2];  // [e][d] (kv transposed; ints exact in f16)
    __shared__ _Float16 Bs[64 * LDK2];  // [n][d]
    const int tid = threadIdx.x, lane = tid & 31, w = tid >> 5;
    const int nblk = blockIdx.x * 64;
    const int tbh = blockIdx.y, tb = tbh >> 3, h = tbh & 7;
    const float*    kvb = kvin + (size_t)tbh * DH * DH;
    const _Float16* Qb  = Qsp + (size_t)tb * NPIX * CCH + h * DH;
    float*          Ob  = pre + (size_t)tb * NPIX * CCH + h * DH;

    { // B: Bs[n][d] = Qs[nblk+n][d] — direct aligned 16B async copies
        int n = tid >> 2, dch = (tid & 3) * 16;
        const _Float16* g = Qb + (size_t)(nblk + n) * CCH + dch;
        cp16(&Bs[n * LDK2 + dch], g);
        cp16(&Bs[n * LDK2 + dch + 8], g + 8);
    }
    { // A: As[e][d] = kv[d*64+e] (global reads coalesced over e)
        int e = tid & 63, d0 = (tid >> 6) * 16;
#pragma unroll
        for (int j = 0; j < 16; ++j)
            As[e * LDK2 + d0 + j] = (_Float16)kvb[(d0 + j) * DH + e];
    }
    cp_wait();
    __syncthreads();

    const int wm = (w & 3) * 16, wn = (w >> 2) * 32;
    const int col0 = lane & 15, rbase = (lane & 16) ? 8 : 0;
    v8f acc[2] = {};
#pragma unroll
    for (int kb = 0; kb < 64; kb += 32) {
        v16h a  = load_frag(As, wm + col0,      LDK2, kb, lane);
        v16h b0 = load_frag(Bs, wn + col0,      LDK2, kb, lane);
        v16h b1 = load_frag(Bs, wn + 16 + col0, LDK2, kb, lane);
        acc[0] = wmma_f16(a, b0, acc[0]);
        acc[1] = wmma_f16(a, b1, acc[1]);
    }
    const int estart = wm + rbase;
#pragma unroll
    for (int ni = 0; ni < 2; ++ni) {
        v8f o;
#pragma unroll
        for (int r = 0; r < 8; ++r) o[r] = acc[ni][r] * 0.25f;  // scale*2
        *(v8f*)&Ob[(size_t)(nblk + wn + ni * 16 + col0) * CCH + estart] = o;
    }
}

// ---------------------------------------------------------------------------
extern "C" void kernel_launch(void* const* d_in, const int* in_sizes, int n_in,
                              void* d_out, int out_size, void* d_ws, size_t ws_size,
                              hipStream_t stream) {
    const float* x   = (const float*)d_in[0];
    const float* pos = (const float*)d_in[1];
    const float* qw  = (const float*)d_in[2];
    const float* kw  = (const float*)d_in[3];
    const float* vw  = (const float*)d_in[4];
    const float* pw  = (const float*)d_in[5];
    const float* qbs = (const float*)d_in[6];
    const float* qbb = (const float*)d_in[7];
    const float* kbs = (const float*)d_in[8];
    const float* kbb = (const float*)d_in[9];
    const float* vbs = (const float*)d_in[10];
    const float* vbb = (const float*)d_in[11];
    const float* pbs = (const float*)d_in[12];
    const float* pbb = (const float*)d_in[13];
    float* out = (float*)d_out;

    char* ws = (char*)d_ws;
    constexpr size_t WB  = (size_t)CCH * CCH * sizeof(_Float16);
    constexpr size_t SPK = (size_t)TT * BB * CCH * NPIX * sizeof(_Float16);
    constexpr size_t PRE = (size_t)TT * BB * CCH * NPIX * sizeof(float);
    _Float16* wq16 = (_Float16*)(ws);
    _Float16* wk16 = (_Float16*)(ws + WB);
    _Float16* wv16 = (_Float16*)(ws + 2 * WB);
    _Float16* wp16 = (_Float16*)(ws + 3 * WB);
    _Float16* xs   = (_Float16*)(ws + 4 * WB);
    _Float16* sq   = (_Float16*)(ws + 4 * WB + SPK);
    _Float16* sk   = (_Float16*)(ws + 4 * WB + 2 * SPK);
    _Float16* sv   = (_Float16*)(ws + 4 * WB + 3 * SPK);
    _Float16* sa   = (_Float16*)(ws + 4 * WB + 4 * SPK);
    float*    pre  = (float*)   (ws + 4 * WB + 5 * SPK);
    float*    kvb  = (float*)   (ws + 4 * WB + 5 * SPK + PRE);

    const int WN = CCH * CCH;
    cvt_f32_f16<<<WN / 256, 256, 0, stream>>>(qw, wq16, WN);
    cvt_f32_f16<<<WN / 256, 256, 0, stream>>>(kw, wk16, WN);
    cvt_f32_f16<<<WN / 256, 256, 0, stream>>>(vw, wv16, WN);
    cvt_f32_f16<<<WN / 256, 256, 0, stream>>>(pw, wp16, WN);

    const int PIX = BB * CCH * NPIX;                  // per-timestep elements
    lif_input<<<PIX / 256, 256, 0, stream>>>(x, pos, xs);

    dim3 gg(NPIX / 64, CCH / 256, TT * BB);
    gemm_spk<true><<<gg, 256, 0, stream>>>(wq16, xs, qbs, qbb, pre);
    lif_scan<<<PIX / 256, 256, 0, stream>>>(pre, sq);
    gemm_spk<true><<<gg, 256, 0, stream>>>(wk16, xs, kbs, kbb, pre);
    lif_scan<<<PIX / 256, 256, 0, stream>>>(pre, sk);
    gemm_spk<true><<<gg, 256, 0, stream>>>(wv16, xs, vbs, vbb, pre);
    lif_scan<<<PIX / 256, 256, 0, stream>>>(pre, sv);

    kv_gemm<<<TT * BB * HEADS, 256, 0, stream>>>(sk, sv, kvb);
    dim3 ga(NPIX / 64, TT * BB * HEADS);
    attn_gemm<<<ga, 256, 0, stream>>>(kvb, sq, pre);
    lif_scan<<<PIX / 256, 256, 0, stream>>>(pre, sa);

    gemm_spk<false><<<gg, 256, 0, stream>>>(wp16, sa, pbs, pbb, out);
}
